// ISDALoss_45526653338014
// MI455X (gfx1250) — matrix operand order repaired and checked by reading, not compiled, not run on hardware
//
#include <hip/hip_runtime.h>
#include <cstdint>
#include <math.h>

// ISDA loss, fused for MI455X (gfx1250, wave32, WMMA).
// Refactor: sigma2 = ratio * ( [cv_k | -2*Wk*cv_k] @ [W^2 | W]^T + t3[n] )
// -> single bf16 GEMM, K = 2*A = 1536, f32 accumulation, fused softmax/NLL.

#define NTOT 8192
#define CDIM 1000
#define CPAD 1024
#define ADIM 768
#define KDIM 1536   // 2*ADIM

typedef __attribute__((ext_vector_type(16))) __bf16 v16bf;
typedef __attribute__((ext_vector_type(8)))  float  v8f;

__device__ __forceinline__ unsigned short f32_to_bf16_rne(float f) {
    unsigned int u = __float_as_uint(f);
    unsigned int r = u + 0x7FFFu + ((u >> 16) & 1u);   // round-to-nearest-even
    return (unsigned short)(r >> 16);
}

// ---------------------------------------------------------------------------
// Prep 1: B-operand, transposed + padded:  btt[c][k], c in [0,1024), k in [0,1536)
//         k < 768 -> W[c][k]^2 ; k >= 768 -> W[c][k-768] ; zero for c >= 1000.
// ---------------------------------------------------------------------------
__global__ void isda_build_bt(const float* __restrict__ W,
                              unsigned short* __restrict__ btt) {
    const int c = blockIdx.x;                       // 0..CPAD-1
    for (int a = threadIdx.x; a < ADIM; a += blockDim.x) {
        float w = (c < CDIM) ? W[(size_t)c * ADIM + a] : 0.0f;
        btt[(size_t)c * KDIM + a]        = f32_to_bf16_rne(w * w);
        btt[(size_t)c * KDIM + ADIM + a] = f32_to_bf16_rne(w);
    }
}

// ---------------------------------------------------------------------------
// Prep 2: per-sample gather -> packedA[n][0:768]=cv_k, [768:1536]=-2*Wk*cv_k (bf16)
//         and t3[n] = sum_a Wk^2 * cv_k  (f32)
// ---------------------------------------------------------------------------
__global__ void isda_gather_pack(const float* __restrict__ W,
                                 const float* __restrict__ CV,
                                 const int*   __restrict__ labels,
                                 unsigned short* __restrict__ packedA,
                                 float* __restrict__ t3) {
    const int n = blockIdx.x;
    const int l = labels[n];
    float part = 0.0f;
    for (int a = threadIdx.x; a < ADIM; a += blockDim.x) {
        float w  = W [(size_t)l * ADIM + a];
        float cv = CV[(size_t)l * ADIM + a];
        packedA[(size_t)n * KDIM + a]        = f32_to_bf16_rne(cv);
        packedA[(size_t)n * KDIM + ADIM + a] = f32_to_bf16_rne(-2.0f * w * cv);
        part += w * w * cv;
    }
    // wave32 + cross-wave reduction
    #pragma unroll
    for (int off = 16; off; off >>= 1) part += __shfl_xor(part, off, 32);
    __shared__ float wsum[8];
    const int lane = threadIdx.x & 31, wv = threadIdx.x >> 5;
    if (lane == 0) wsum[wv] = part;
    __syncthreads();
    if (threadIdx.x == 0) {
        float s = 0.0f;
        #pragma unroll
        for (int i = 0; i < 8; i++) s += wsum[i];
        t3[n] = s;
    }
}

// ---------------------------------------------------------------------------
// Fused GEMM + epilogue + softmax + per-row NLL.
// Block: 256 threads = 8 wave32, owns 16 rows x 1024 cols.
// Wave w: column tiles ct = w*16 + j*128, j = 0..7.  K loop: 48 x (K=32) WMMA.
// ---------------------------------------------------------------------------
__global__ __launch_bounds__(256)
void isda_fused(const unsigned short* __restrict__ packedA,
                const unsigned short* __restrict__ btt,
                const float* __restrict__ logits,
                const int*   __restrict__ labels,
                const float* __restrict__ ratio_p,
                const float* __restrict__ t3g,
                float* __restrict__ rowloss) {
    __shared__ float augS[16 * CPAD];               // 64 KB augmented-logit tile

    const int lane = threadIdx.x & 31;
    const int wv   = threadIdx.x >> 5;              // 0..7
    const int n0   = blockIdx.x * 16;
    const int rA   = lane & 15;
    const int hi   = lane >> 4;                     // half-wave select
    const float ratio = ratio_p[0];

    union ABFrag { uint4 q[2]; v16bf v; };

    v8f acc[8] = {};                                // 64 f32 accumulators / lane

    // A fragment base: row = n0+rA, per-lane K chunks at hi*8 and hi*8+16 (ISA layout)
    const unsigned short* aRow = packedA + (size_t)(n0 + rA) * KDIM + hi * 8;

    for (int k0 = 0; k0 < KDIM; k0 += 32) {
        ABFrag a;
        a.q[0] = *(const uint4*)(aRow + k0);        // K = k0+hi*8   .. +7
        a.q[1] = *(const uint4*)(aRow + k0 + 16);   // K = k0+hi*8+16.. +23
        #pragma unroll
        for (int j = 0; j < 8; j++) {
            const int ct = wv * 16 + j * 128;
            // B fragment: col = ct+rA, K chunks at k0 + hi*8 and k0 + hi*8 + 16 (pre-transposed)
            const unsigned short* bp = btt + (size_t)(ct + rA) * KDIM + k0 + hi * 8;
            ABFrag b;
            b.q[0] = *(const uint4*)(bp);
            b.q[1] = *(const uint4*)(bp + 16);
            acc[j] = __builtin_amdgcn_wmma_f32_16x16x32_bf16(
                         false, a.v, false, b.v, (short)0, acc[j], false, false);
        }
    }

    // Epilogue: aug = logits + 0.5*ratio*(acc + t3);  pad columns -> -inf
    #pragma unroll
    for (int j = 0; j < 8; j++) {
        const int col = wv * 16 + j * 128 + rA;
        #pragma unroll
        for (int v = 0; v < 8; v++) {
            const int rowM = hi * 8 + v;            // C/D layout: VGPR v, half-wave hi
            float val;
            if (col < CDIM) {
                const float t3v = t3g[n0 + rowM];
                val = logits[(size_t)(n0 + rowM) * CDIM + col]
                      + 0.5f * ratio * (acc[j][v] + t3v);
            } else {
                val = -INFINITY;
            }
            augS[rowM * CPAD + col] = val;
        }
    }
    __syncthreads();

    // Softmax + NLL: wave w handles rows w and w+8 (wave32 shfl reductions)
    #pragma unroll
    for (int rr = 0; rr < 2; rr++) {
        const int r = wv + rr * 8;
        float m = -INFINITY;
        for (int c = lane; c < CPAD; c += 32) m = fmaxf(m, augS[r * CPAD + c]);
        #pragma unroll
        for (int off = 16; off; off >>= 1) m = fmaxf(m, __shfl_xor(m, off, 32));
        float s = 0.0f;
        for (int c = lane; c < CPAD; c += 32) s += __expf(augS[r * CPAD + c] - m);
        #pragma unroll
        for (int off = 16; off; off >>= 1) s += __shfl_xor(s, off, 32);
        if (lane == 0) {
            const int l = labels[n0 + r];
            rowloss[n0 + r] = m + __logf(s) - augS[r * CPAD + l];
        }
    }
}

// ---------------------------------------------------------------------------
// Deterministic final reduction: mean over N row losses (no float atomics).
// ---------------------------------------------------------------------------
__global__ void isda_reduce(const float* __restrict__ rowloss, float* __restrict__ out) {
    __shared__ float sm[256];
    float s = 0.0f;
    for (int i = threadIdx.x; i < NTOT; i += 256) s += rowloss[i];
    sm[threadIdx.x] = s;
    __syncthreads();
    for (int off = 128; off; off >>= 1) {
        if (threadIdx.x < off) sm[threadIdx.x] += sm[threadIdx.x + off];
        __syncthreads();
    }
    if (threadIdx.x == 0) out[0] = sm[0] / (float)NTOT;
}

extern "C" void kernel_launch(void* const* d_in, const int* in_sizes, int n_in,
                              void* d_out, int out_size, void* d_ws, size_t ws_size,
                              hipStream_t stream) {
    const float* W      = (const float*)d_in[0];   // fc_weight [C, A]
    // d_in[1] = features (dead input in the reference math)
    const float* logits = (const float*)d_in[2];   // [N, C]
    const int*   labels = (const int*)  d_in[3];   // [N]
    const float* ratio  = (const float*)d_in[4];   // scalar
    const float* CV     = (const float*)d_in[5];   // cv_matrix [C, A]
    // d_in[6] = manner (unused)

    // Workspace layout (~28.4 MB total)
    char* ws = (char*)d_ws;
    unsigned short* btt     = (unsigned short*)ws;                       // 1024*1536*2 = 3 MB
    unsigned short* packedA = (unsigned short*)(ws + (size_t)CPAD * KDIM * 2);   // 8192*1536*2 = 24 MB
    float* t3      = (float*)(ws + (size_t)CPAD * KDIM * 2 + (size_t)NTOT * KDIM * 2);
    float* rowloss = t3 + NTOT;

    isda_build_bt   <<<CPAD, 256, 0, stream>>>(W, btt);
    isda_gather_pack<<<NTOT, 256, 0, stream>>>(W, CV, labels, packedA, t3);
    isda_fused      <<<NTOT / 16, 256, 0, stream>>>(packedA, btt, logits, labels,
                                                    ratio, t3, rowloss);
    isda_reduce     <<<1, 256, 0, stream>>>(rowloss, (float*)d_out);
}
